// GeneConvLayer_26594437496992
// MI455X (gfx1250) — compile-verified, matrix operand order checked by reading, not run on hardware
//
#include <hip/hip_runtime.h>

typedef __attribute__((ext_vector_type(16))) __bf16 v16bf;
typedef __attribute__((ext_vector_type(8)))  __bf16 v8bf;
typedef __attribute__((ext_vector_type(8)))  float  v8f;

#define N_NODES 50000
#define N_EDGES 1600000
#define DIM     128
#define EDIM    64
#define EPS     1e-5f

#define EDGE_WAVES 2   // waves per block; each wave owns 2 x 16 = 32 edges
#define NODE_WAVES 1   // each wave owns 2 x 16 = 32 nodes

// ---------------- WMMA helpers ----------------

__device__ __forceinline__ v8f wmma_bf(v16bf a, v16bf b, v8f c) {
  // D = A(16x32 bf16) * B(32x16 bf16) + C(16x16 f32)
  return __builtin_amdgcn_wmma_f32_16x16x32_bf16(false, a, false, b, (short)0, c,
                                                 false, false);
}

union V16U { v16bf v; v8bf h[2]; };

// A fragment from a row-major bf16 row: lane holds row M=lane%16, two contiguous
// 8-element K runs at off and off+16 (off = kc*32 + half*8).
__device__ __forceinline__ v16bf a_load(const __bf16* row, int off) {
  V16U u;
  u.h[0] = *(const v8bf*)(row + off);
  u.h[1] = *(const v8bf*)(row + off + 16);
  return u.v;
}

// Same A pattern, converting fp32 source on the fly.
__device__ __forceinline__ v16bf a_cvt_f32(const float* p) {
  v16bf a;
#pragma unroll
  for (int i = 0; i < 8; ++i) { a[i] = (__bf16)p[i]; a[i + 8] = (__bf16)p[16 + i]; }
  return a;
}

// Pre-swizzled B fragment: contiguous 32 bytes per lane.
__device__ __forceinline__ v16bf b_load(const __bf16* frags, int idx, int lane) {
  return *(const v16bf*)(frags + ((size_t)idx * 32 + lane) * 16);
}

// ---------------- utility kernels ----------------

__global__ void zero_kernel(float* p, long long n) {
  long long i = (long long)blockIdx.x * blockDim.x + threadIdx.x;
  if (i < n) p[i] = 0.0f;
}

__global__ void cvt_kernel(const float* __restrict__ x, __bf16* __restrict__ o,
                           long long n) {
  long long i = (long long)blockIdx.x * blockDim.x + threadIdx.x;
  if (i < n) o[i] = (__bf16)x[i];
}

// Swizzle row-major W[fan_in][fan_out] (f32) into per-fragment bf16 B layout.
// grid.x = kchunks*ntiles, block = 32.
__global__ void swizzle_kernel(const float* __restrict__ W, int fan_out, int ntiles,
                               __bf16* __restrict__ out) {
  int frag = blockIdx.x;
  int kc = frag / ntiles, nt = frag % ntiles;
  int lane = threadIdx.x;
  int half = lane >> 4, mr = lane & 15;
  int n = nt * 16 + mr;
  __bf16* o = out + ((size_t)frag * 32 + lane) * 16;
#pragma unroll
  for (int i = 0; i < 16; ++i) {
    int v = i >> 1, pos = i & 1;
    int k = kc * 32 + ((v < 4) ? (half * 8 + v * 2 + pos)
                               : (16 + half * 8 + (v - 4) * 2 + pos));
    o[i] = (__bf16)W[(size_t)k * fan_out + n];
  }
}

// ---------------- edge kernel: edge MLP + LN + message MLP + scatter ----------------
// Each wave processes two 16-edge M-tiles so every B fragment feeds 2 WMMAs.
// B fragments are software double-buffered: fragment i+1 is loaded before the
// WMMAs consuming fragment i, so load latency overlaps matrix work.

__global__ __launch_bounds__(EDGE_WAVES * 32) void edge_kernel(
    const float* __restrict__ eattr, const long long* __restrict__ ei,
    const __bf16* __restrict__ xbf,
    const __bf16* __restrict__ We1f, const float* __restrict__ be1,
    const __bf16* __restrict__ We2f, const float* __restrict__ be2,
    const __bf16* __restrict__ We3f, const float* __restrict__ be3,
    const float* __restrict__ eg, const float* __restrict__ ebeta,
    const __bf16* __restrict__ Wm1f, const float* __restrict__ bm1,
    const __bf16* __restrict__ Wm2f, const float* __restrict__ bm2,
    const __bf16* __restrict__ Wm3f, const float* __restrict__ bm3,
    float* __restrict__ eout, float* __restrict__ msum, float* __restrict__ cnt) {
  __shared__ __attribute__((aligned(16))) __bf16 s_ea[EDGE_WAVES][32][64];
  __shared__ __attribute__((aligned(16))) float  s_tmp[EDGE_WAVES][32][68];
  __shared__ __attribute__((aligned(16))) __bf16 s_h[EDGE_WAVES][32][128];
  __shared__ int s_dst[EDGE_WAVES][32];

  const int w = threadIdx.x >> 5;
  const int lane = threadIdx.x & 31;
  const int half = lane >> 4;
  const int mr = lane & 15;
  const long long wbase = (long long)(blockIdx.x * EDGE_WAVES + w) * 32;

  const __bf16* xs[2];
  const __bf16* xd[2];
  const float*  er[2];
#pragma unroll
  for (int m = 0; m < 2; ++m) {
    long long e = wbase + m * 16 + mr;
    if (e >= N_EDGES) e = N_EDGES - 1;
    int src = (int)ei[e];
    int dst = (int)ei[(long long)N_EDGES + e];
    if (lane < 16) s_dst[w][m * 16 + mr] = dst;
    xs[m] = xbf + (size_t)src * DIM;
    xd[m] = xbf + (size_t)dst * DIM;
    er[m] = eattr + (size_t)e * EDIM;
  }

  // ---- edge MLP layer 1: [32 x 320] @ [320 x 64]
  v8f e1[2][4] = {};
  {
    v16bf b = b_load(We1f, 0, lane);
#pragma unroll
    for (int kc = 0; kc < 10; ++kc) {
      v16bf a[2];
#pragma unroll
      for (int m = 0; m < 2; ++m) {
        if (kc < 4)      a[m] = a_load(xs[m], kc * 32 + half * 8);
        else if (kc < 6) a[m] = a_cvt_f32(er[m] + (kc - 4) * 32 + half * 8);
        else             a[m] = a_load(xd[m], (kc - 6) * 32 + half * 8);
      }
#pragma unroll
      for (int t = 0; t < 4; ++t) {
        int idx = kc * 4 + t;
        v16bf bn = b_load(We1f, idx < 39 ? idx + 1 : 39, lane);
        e1[0][t] = wmma_bf(a[0], b, e1[0][t]);
        e1[1][t] = wmma_bf(a[1], b, e1[1][t]);
        b = bn;
      }
    }
  }
#pragma unroll
  for (int t = 0; t < 4; ++t) {
    float bs = be1[t * 16 + mr];
#pragma unroll
    for (int m = 0; m < 2; ++m)
#pragma unroll
      for (int r = 0; r < 8; ++r)
        s_h[w][m * 16 + r + 8 * half][t * 16 + mr] =
            (__bf16)fmaxf(e1[m][t][r] + bs, 0.0f);
  }
  __syncthreads();

  // ---- edge MLP layer 2: [32 x 64] @ [64 x 64]
  v8f e2[2][4] = {};
  {
    v16bf b = b_load(We2f, 0, lane);
#pragma unroll
    for (int kc = 0; kc < 2; ++kc) {
      v16bf a0 = a_load(&s_h[w][mr][0],      kc * 32 + half * 8);
      v16bf a1 = a_load(&s_h[w][16 + mr][0], kc * 32 + half * 8);
#pragma unroll
      for (int t = 0; t < 4; ++t) {
        int idx = kc * 4 + t;
        v16bf bn = b_load(We2f, idx < 7 ? idx + 1 : 7, lane);
        e2[0][t] = wmma_bf(a0, b, e2[0][t]);
        e2[1][t] = wmma_bf(a1, b, e2[1][t]);
        b = bn;
      }
    }
  }
#pragma unroll
  for (int t = 0; t < 4; ++t) {
    float bs = be2[t * 16 + mr];
#pragma unroll
    for (int m = 0; m < 2; ++m)
#pragma unroll
      for (int r = 0; r < 8; ++r)
        s_h[w][m * 16 + r + 8 * half][64 + t * 16 + mr] =
            (__bf16)fmaxf(e2[m][t][r] + bs, 0.0f);
  }
  __syncthreads();

  // ---- edge MLP layer 3: [32 x 64] @ [64 x 64] (no relu)
  v8f e3[2][4] = {};
  {
    v16bf b = b_load(We3f, 0, lane);
#pragma unroll
    for (int kc = 0; kc < 2; ++kc) {
      v16bf a0 = a_load(&s_h[w][mr][64],      kc * 32 + half * 8);
      v16bf a1 = a_load(&s_h[w][16 + mr][64], kc * 32 + half * 8);
#pragma unroll
      for (int t = 0; t < 4; ++t) {
        int idx = kc * 4 + t;
        v16bf bn = b_load(We3f, idx < 7 ? idx + 1 : 7, lane);
        e3[0][t] = wmma_bf(a0, b, e3[0][t]);
        e3[1][t] = wmma_bf(a1, b, e3[1][t]);
        b = bn;
      }
    }
  }
#pragma unroll
  for (int t = 0; t < 4; ++t) {
    float bs = be3[t * 16 + mr];
#pragma unroll
    for (int m = 0; m < 2; ++m)
#pragma unroll
      for (int r = 0; r < 8; ++r)
        s_tmp[w][m * 16 + r + 8 * half][t * 16 + mr] = e3[m][t][r] + bs;
  }
  __syncthreads();

  // ---- residual + LayerNorm over 64 features -> edge_attr_new
#pragma unroll
  for (int m = 0; m < 2; ++m) {
    int row = m * 16 + mr;
    float s = 0.f, ss = 0.f;
#pragma unroll
    for (int j0 = 0; j0 < 32; ++j0) {
      int j = half * 32 + j0;
      float y = er[m][j] + s_tmp[w][row][j];
      s_tmp[w][row][j] = y;
      s += y; ss += y * y;
    }
    s  += __shfl_xor(s, 16);
    ss += __shfl_xor(ss, 16);
    float mean = s * (1.0f / 64);
    float var  = ss * (1.0f / 64) - mean * mean;
    float rstd = rsqrtf(var + EPS);
    bool ok = (wbase + row) < N_EDGES;
#pragma unroll
    for (int j0 = 0; j0 < 32; ++j0) {
      int j = half * 32 + j0;
      float o = (s_tmp[w][row][j] - mean) * rstd * eg[j] + ebeta[j];
      s_ea[w][row][j] = (__bf16)o;
      if (ok) eout[(size_t)(wbase + row) * EDIM + j] = o;
    }
  }
  __syncthreads();

  // ---- message MLP layer 1: [32 x 320] @ [320 x 128]
  v8f m1[2][8] = {};
  {
    v16bf b = b_load(Wm1f, 0, lane);
#pragma unroll
    for (int kc = 0; kc < 10; ++kc) {
      v16bf a[2];
#pragma unroll
      for (int m = 0; m < 2; ++m) {
        if (kc < 4)      a[m] = a_load(xs[m], kc * 32 + half * 8);
        else if (kc < 6) a[m] = a_load(&s_ea[w][m * 16 + mr][0],
                                       (kc - 4) * 32 + half * 8);
        else             a[m] = a_load(xd[m], (kc - 6) * 32 + half * 8);
      }
#pragma unroll
      for (int t = 0; t < 8; ++t) {
        int idx = kc * 8 + t;
        v16bf bn = b_load(Wm1f, idx < 79 ? idx + 1 : 79, lane);
        m1[0][t] = wmma_bf(a[0], b, m1[0][t]);
        m1[1][t] = wmma_bf(a[1], b, m1[1][t]);
        b = bn;
      }
    }
  }
#pragma unroll
  for (int t = 0; t < 8; ++t) {
    float bs = bm1[t * 16 + mr];
#pragma unroll
    for (int m = 0; m < 2; ++m)
#pragma unroll
      for (int r = 0; r < 8; ++r)
        s_h[w][m * 16 + r + 8 * half][t * 16 + mr] =
            (__bf16)fmaxf(m1[m][t][r] + bs, 0.0f);
  }
  __syncthreads();

  // ---- message MLP layer 2: [32 x 128] @ [128 x 128]
  v8f m2[2][8] = {};
  __bf16* tb = (__bf16*)&s_tmp[w][0][0];  // 32 rows, stride 136 bf16
  {
    v16bf b = b_load(Wm2f, 0, lane);
#pragma unroll
    for (int kc = 0; kc < 4; ++kc) {
      v16bf a0 = a_load(&s_h[w][mr][0],      kc * 32 + half * 8);
      v16bf a1 = a_load(&s_h[w][16 + mr][0], kc * 32 + half * 8);
#pragma unroll
      for (int t = 0; t < 8; ++t) {
        int idx = kc * 8 + t;
        v16bf bn = b_load(Wm2f, idx < 31 ? idx + 1 : 31, lane);
        m2[0][t] = wmma_bf(a0, b, m2[0][t]);
        m2[1][t] = wmma_bf(a1, b, m2[1][t]);
        b = bn;
      }
    }
  }
#pragma unroll
  for (int t = 0; t < 8; ++t) {
    float bs = bm2[t * 16 + mr];
#pragma unroll
    for (int m = 0; m < 2; ++m)
#pragma unroll
      for (int r = 0; r < 8; ++r)
        tb[(m * 16 + r + 8 * half) * 136 + t * 16 + mr] =
            (__bf16)fmaxf(m2[m][t][r] + bs, 0.0f);
  }
  __syncthreads();

  // ---- message MLP layer 3: [32 x 128] @ [128 x 128] (no relu)
  v8f m3[2][8] = {};
  {
    v16bf b = b_load(Wm3f, 0, lane);
#pragma unroll
    for (int kc = 0; kc < 4; ++kc) {
      v16bf a0 = a_load(tb + mr * 136,        kc * 32 + half * 8);
      v16bf a1 = a_load(tb + (16 + mr) * 136, kc * 32 + half * 8);
#pragma unroll
      for (int t = 0; t < 8; ++t) {
        int idx = kc * 8 + t;
        v16bf bn = b_load(Wm3f, idx < 31 ? idx + 1 : 31, lane);
        m3[0][t] = wmma_bf(a0, b, m3[0][t]);
        m3[1][t] = wmma_bf(a1, b, m3[1][t]);
        b = bn;
      }
    }
  }

  // ---- scatter-sum at dst + per-edge counts
  if (lane < 16) {
#pragma unroll
    for (int m = 0; m < 2; ++m)
      if ((wbase + m * 16 + mr) < N_EDGES)
        atomicAdd(&cnt[s_dst[w][m * 16 + mr]], 1.0f);
  }
#pragma unroll
  for (int t = 0; t < 8; ++t) {
    float bs = bm3[t * 16 + mr];
#pragma unroll
    for (int m = 0; m < 2; ++m)
#pragma unroll
      for (int r = 0; r < 8; ++r) {
        int row = m * 16 + r + 8 * half;
        if (wbase + row < N_EDGES)
          atomicAdd(&msum[(size_t)s_dst[w][row] * DIM + t * 16 + mr],
                    m3[m][t][r] + bs);
      }
  }
}

// ---------------- node kernel: mean + LN + FFN + LN ----------------
// One wave per block, two 16-node M-tiles per wave.

__global__ __launch_bounds__(NODE_WAVES * 32) void node_kernel(
    const float* __restrict__ x, const float* __restrict__ msum,
    const float* __restrict__ cnt,
    const float* __restrict__ g0, const float* __restrict__ b0,
    const __bf16* __restrict__ Wf1f, const float* __restrict__ bf1,
    const __bf16* __restrict__ Wf2f, const float* __restrict__ bf2,
    const float* __restrict__ g1, const float* __restrict__ b1,
    float* __restrict__ xout) {
  __shared__ __attribute__((aligned(16))) __bf16 s_xb[32][128];
  __shared__ __attribute__((aligned(16))) __bf16 s_hh[32][512];
  __shared__ __attribute__((aligned(16))) float  s_f[32][132];

  const int lane = threadIdx.x & 31;
  const int half = lane >> 4;
  const int mr = lane & 15;
  const long long nbase = (long long)blockIdx.x * 32;

  long long nrow[2];
  bool ok[2];
#pragma unroll
  for (int m = 0; m < 2; ++m) {
    nrow[m] = nbase + m * 16 + mr;
    ok[m] = nrow[m] < N_NODES;
    if (!ok[m]) nrow[m] = N_NODES - 1;
  }

  // ---- x + mean-aggregated message, LayerNorm over 128
#pragma unroll
  for (int m = 0; m < 2; ++m) {
    int row = m * 16 + mr;
    const float* xr = x + (size_t)nrow[m] * DIM;
    const float* ms = msum + (size_t)nrow[m] * DIM;
    const float inv = 1.0f / fmaxf(cnt[nrow[m]], 1.0f);
    float s = 0.f, ss = 0.f;
#pragma unroll
    for (int j0 = 0; j0 < 64; ++j0) {
      int j = half * 64 + j0;
      float y = xr[j] + ms[j] * inv;
      s_f[row][j] = y;
      s += y; ss += y * y;
    }
    s += __shfl_xor(s, 16); ss += __shfl_xor(ss, 16);
    float mean = s * (1.0f / 128), var = ss * (1.0f / 128) - mean * mean;
    float rstd = rsqrtf(var + EPS);
#pragma unroll
    for (int j0 = 0; j0 < 64; ++j0) {
      int j = half * 64 + j0;
      float o = (s_f[row][j] - mean) * rstd * g0[j] + b0[j];
      s_f[row][j] = o;            // fp32 residual copy
      s_xb[row][j] = (__bf16)o;   // bf16 A operand
    }
  }
  __syncthreads();

  // ---- FFN layer 1: [32 x 128] @ [128 x 512], relu
#pragma unroll
  for (int g = 0; g < 4; ++g) {
    v8f a1[2][8] = {};
    {
      v16bf b = b_load(Wf1f, g * 8, lane);
#pragma unroll
      for (int kc = 0; kc < 4; ++kc) {
        v16bf a0  = a_load(&s_xb[mr][0],      kc * 32 + half * 8);
        v16bf a1v = a_load(&s_xb[16 + mr][0], kc * 32 + half * 8);
#pragma unroll
        for (int t = 0; t < 8; ++t) {
          int nidx = (t < 7) ? (kc * 32 + g * 8 + t + 1)
                   : (kc < 3 ? ((kc + 1) * 32 + g * 8)
                             : (kc * 32 + g * 8 + t));
          v16bf bn = b_load(Wf1f, nidx, lane);
          a1[0][t] = wmma_bf(a0,  b, a1[0][t]);
          a1[1][t] = wmma_bf(a1v, b, a1[1][t]);
          b = bn;
        }
      }
    }
#pragma unroll
    for (int t = 0; t < 8; ++t) {
      int col = (g * 8 + t) * 16 + mr;
      float bs = bf1[col];
#pragma unroll
      for (int m = 0; m < 2; ++m)
#pragma unroll
        for (int r = 0; r < 8; ++r)
          s_hh[m * 16 + r + 8 * half][col] =
              (__bf16)fmaxf(a1[m][t][r] + bs, 0.0f);
    }
  }
  __syncthreads();

  // ---- FFN layer 2: [32 x 512] @ [512 x 128]
  v8f a2[2][8] = {};
  {
    v16bf b = b_load(Wf2f, 0, lane);
#pragma unroll
    for (int kc = 0; kc < 16; ++kc) {
      v16bf a0 = a_load(&s_hh[mr][0],      kc * 32 + half * 8);
      v16bf a1 = a_load(&s_hh[16 + mr][0], kc * 32 + half * 8);
#pragma unroll
      for (int t = 0; t < 8; ++t) {
        int idx = kc * 8 + t;
        v16bf bn = b_load(Wf2f, idx < 127 ? idx + 1 : 127, lane);
        a2[0][t] = wmma_bf(a0, b, a2[0][t]);
        a2[1][t] = wmma_bf(a1, b, a2[1][t]);
        b = bn;
      }
    }
  }
  __syncthreads();
  float* fs = (float*)&s_hh[0][0];  // reuse as 32 x 128 f32 staging
#pragma unroll
  for (int t = 0; t < 8; ++t) {
    int col = t * 16 + mr;
    float bs = bf2[col];
#pragma unroll
    for (int m = 0; m < 2; ++m)
#pragma unroll
      for (int r = 0; r < 8; ++r) {
        int row = m * 16 + r + 8 * half;
        fs[row * DIM + col] = a2[m][t][r] + bs + s_f[row][col];
      }
  }
  __syncthreads();

  // ---- final LayerNorm over 128
#pragma unroll
  for (int m = 0; m < 2; ++m) {
    int row = m * 16 + mr;
    float s = 0.f, ss = 0.f;
#pragma unroll
    for (int j0 = 0; j0 < 64; ++j0) {
      int j = half * 64 + j0;
      float y = fs[row * DIM + j];
      s += y; ss += y * y;
    }
    s += __shfl_xor(s, 16); ss += __shfl_xor(ss, 16);
    float mean = s * (1.0f / 128), var = ss * (1.0f / 128) - mean * mean;
    float rstd = rsqrtf(var + EPS);
    if (ok[m]) {
#pragma unroll
      for (int j0 = 0; j0 < 64; ++j0) {
        int j = half * 64 + j0;
        float y = fs[row * DIM + j];
        xout[(size_t)nrow[m] * DIM + j] = (y - mean) * rstd * g1[j] + b1[j];
      }
    }
  }
}

// ---------------- host launch ----------------

extern "C" void kernel_launch(void* const* d_in, const int* in_sizes, int n_in,
                              void* d_out, int out_size, void* d_ws, size_t ws_size,
                              hipStream_t stream) {
  (void)in_sizes; (void)n_in; (void)out_size; (void)ws_size;

  const float*     x     = (const float*)d_in[0];
  const long long* ei    = (const long long*)d_in[1];
  const float*     eattr = (const float*)d_in[2];
  const float* We1 = (const float*)d_in[3];  const float* be1 = (const float*)d_in[4];
  const float* We2 = (const float*)d_in[5];  const float* be2 = (const float*)d_in[6];
  const float* We3 = (const float*)d_in[7];  const float* be3 = (const float*)d_in[8];
  const float* eg  = (const float*)d_in[9];  const float* ebeta = (const float*)d_in[10];
  const float* Wm1 = (const float*)d_in[11]; const float* bm1 = (const float*)d_in[12];
  const float* Wm2 = (const float*)d_in[13]; const float* bm2 = (const float*)d_in[14];
  const float* Wm3 = (const float*)d_in[15]; const float* bm3 = (const float*)d_in[16];
  const float* g0  = (const float*)d_in[17]; const float* b0  = (const float*)d_in[18];
  const float* Wf1 = (const float*)d_in[19]; const float* bf1 = (const float*)d_in[20];
  const float* Wf2 = (const float*)d_in[21]; const float* bf2 = (const float*)d_in[22];
  const float* g1  = (const float*)d_in[23]; const float* b1  = (const float*)d_in[24];

  float* out_x = (float*)d_out;                       // [N, 128]
  float* out_e = out_x + (size_t)N_NODES * DIM;       // [E, 64]

  char* ws = (char*)d_ws;
  size_t off = 0;
  auto alloc = [&](size_t bytes) -> void* {
    void* p = ws + off;
    off = (off + bytes + 255) & ~(size_t)255;
    return p;
  };
  __bf16* xbf  = (__bf16*)alloc((size_t)N_NODES * DIM * 2);
  float*  msum = (float*) alloc((size_t)N_NODES * DIM * 4);
  float*  cnt  = (float*) alloc((size_t)N_NODES * 4);
  __bf16* We1f = (__bf16*)alloc((size_t)10 * 4 * 512 * 2);
  __bf16* We2f = (__bf16*)alloc((size_t)2 * 4 * 512 * 2);
  __bf16* We3f = (__bf16*)alloc((size_t)2 * 4 * 512 * 2);
  __bf16* Wm1f = (__bf16*)alloc((size_t)10 * 8 * 512 * 2);
  __bf16* Wm2f = (__bf16*)alloc((size_t)4 * 8 * 512 * 2);
  __bf16* Wm3f = (__bf16*)alloc((size_t)4 * 8 * 512 * 2);
  __bf16* Wf1f = (__bf16*)alloc((size_t)4 * 32 * 512 * 2);
  __bf16* Wf2f = (__bf16*)alloc((size_t)16 * 8 * 512 * 2);

  const long long nz = (long long)N_NODES * DIM;
  zero_kernel<<<(int)((nz + 255) / 256), 256, 0, stream>>>(msum, nz);
  zero_kernel<<<(N_NODES + 255) / 256, 256, 0, stream>>>(cnt, N_NODES);
  cvt_kernel<<<(int)((nz + 255) / 256), 256, 0, stream>>>(x, xbf, nz);

  swizzle_kernel<<<10 * 4, 32, 0, stream>>>(We1, 64, 4, We1f);
  swizzle_kernel<<< 2 * 4, 32, 0, stream>>>(We2, 64, 4, We2f);
  swizzle_kernel<<< 2 * 4, 32, 0, stream>>>(We3, 64, 4, We3f);
  swizzle_kernel<<<10 * 8, 32, 0, stream>>>(Wm1, 128, 8, Wm1f);
  swizzle_kernel<<< 4 * 8, 32, 0, stream>>>(Wm2, 128, 8, Wm2f);
  swizzle_kernel<<< 4 * 8, 32, 0, stream>>>(Wm3, 128, 8, Wm3f);
  swizzle_kernel<<< 4 * 32, 32, 0, stream>>>(Wf1, 512, 32, Wf1f);
  swizzle_kernel<<<16 * 8, 32, 0, stream>>>(Wf2, 128, 8, Wf2f);

  const int egrid = N_EDGES / (EDGE_WAVES * 32);  // 1.6M / 64 = 25000, exact
  edge_kernel<<<egrid, EDGE_WAVES * 32, 0, stream>>>(
      eattr, ei, xbf, We1f, be1, We2f, be2, We3f, be3, eg, ebeta,
      Wm1f, bm1, Wm2f, bm2, Wm3f, bm3, out_e, msum, cnt);

  const int ngrid = (N_NODES + 31) / 32;
  node_kernel<<<ngrid, NODE_WAVES * 32, 0, stream>>>(
      x, msum, cnt, g0, b0, Wf1f, bf1, Wf2f, bf2, g1, b1, out_x);
}
